// Attention_17162689314958
// MI455X (gfx1250) — matrix operand order repaired and checked by reading, not compiled, not run on hardware
//
#include <hip/hip_runtime.h>
#include <math.h>

#define BB    32
#define TT    2048
#define QDIM  1024
#define MDIM  512
#define ADIM  1024
#define NEG_INF_F (-1.0e9f)

typedef __bf16        bf16x16 __attribute__((ext_vector_type(16)));
typedef float         f32x8   __attribute__((ext_vector_type(8)));
typedef float         f32x4   __attribute__((ext_vector_type(4)));
typedef unsigned int  u32x4   __attribute__((ext_vector_type(4)));
typedef unsigned int  u32x2   __attribute__((ext_vector_type(2)));

static __device__ __forceinline__ unsigned short f2bf(float f) {
  unsigned int u = __float_as_uint(f);
  u += 0x7FFFu + ((u >> 16) & 1u);          // round-to-nearest-even
  return (unsigned short)(u >> 16);
}

// Hardware tanh (CDNA5 V_TANH_F32 transcendental) instead of ~20-op libcall.
static __device__ __forceinline__ float fast_tanhf(float x) {
#if __has_builtin(__builtin_amdgcn_tanhf)
  return __builtin_amdgcn_tanhf(x);
#elif __has_builtin(__builtin_amdgcn_tanh_f32)
  return __builtin_amdgcn_tanh_f32(x);
#else
  float r;
  // TRANS op: pad with v_nops to honor the 1-op TRANS RAW hazard window.
  asm volatile("v_tanh_f32 %0, %1\n\tv_nop\n\tv_nop" : "=v"(r) : "v"(x));
  return r;
#endif
}

// ---------------------------------------------------------------------------
// Kernel 0: convert Wm [ADIM, MDIM] f32 -> bf16 bits (read once, 2MB -> 1MB)
// ---------------------------------------------------------------------------
__global__ __launch_bounds__(256) void cvt_wm_kernel(
    const float* __restrict__ wm, unsigned short* __restrict__ wmbf, int n) {
  int i = blockIdx.x * 256 + threadIdx.x;
  if (i < n) wmbf[i] = f2bf(wm[i]);
}

// ---------------------------------------------------------------------------
// Kernel 1: q[b,a] = sum_k query[b,k] * Wq[a,k]   (67 MFLOP, plain f32)
// ---------------------------------------------------------------------------
__global__ __launch_bounds__(256) void q_kernel(
    const float* __restrict__ query, const float* __restrict__ wq,
    float* __restrict__ qout) {
  __shared__ float sq[QDIM];
  const int b = blockIdx.y;
  const int a = blockIdx.x * 256 + threadIdx.x;
  for (int i = threadIdx.x; i < QDIM; i += 256)
    sq[i] = query[(size_t)b * QDIM + i];
  __syncthreads();
  const float* wr = wq + (size_t)a * QDIM;
  float acc = 0.0f;
  #pragma unroll 4
  for (int k = 0; k < QDIM; k += 4) {
    f32x4 w = *reinterpret_cast<const f32x4*>(wr + k);
    acc += w[0] * sq[k] + w[1] * sq[k + 1] + w[2] * sq[k + 2] + w[3] * sq[k + 3];
  }
  qout[(size_t)b * ADIM + a] = acc;
}

// ---------------------------------------------------------------------------
// Kernel 2: fused  scores[b,t] = v . tanh(q[b,:] + memory@Wm^T)
// grid (TT/32, BB), 256 threads = 8 waves. Block = 32 tokens x 1024 cols
// (two 16-row M-tiles so every B fragment feeds TWO WMMAs -> halves L2
// traffic on Wm). Wave w owns columns [w*128, w*128+128) as 8 N-tiles.
// K-loop over MDIM in steps of 32 using v_wmma_f32_16x16x32_bf16.
// ---------------------------------------------------------------------------
__global__ __launch_bounds__(256) void score_kernel(
    const float* __restrict__ memory,          // [BB, TT, MDIM] f32
    const unsigned short* __restrict__ wmbf,   // [ADIM, MDIM] bf16 bits
    const float* __restrict__ qmat,            // [BB, ADIM] f32
    const float* __restrict__ vvec,            // [ADIM] f32
    float* __restrict__ scores)                // [BB, TT] f32
{
  __shared__ unsigned short ldsA[32 * 32];     // 32 tokens x 32 K, bf16
  __shared__ float sred[32];

  const int b    = blockIdx.y;
  const int t0   = blockIdx.x * 32;
  const int tid  = threadIdx.x;
  const int wave = tid >> 5;
  const int lane = tid & 31;
  const int h    = lane >> 4;                  // half-wave index
  const int ln   = lane & 15;

  const float* memB = memory + ((size_t)b * TT + t0) * MDIM;

  f32x8 acc0[8], acc1[8];
  #pragma unroll
  for (int j = 0; j < 8; ++j) {
    #pragma unroll
    for (int e = 0; e < 8; ++e) { acc0[j][e] = 0.0f; acc1[j][e] = 0.0f; }
  }

  const int sr = (tid * 4) >> 5;               // staging row (token 0..31)
  const int sc = (tid * 4) & 31;               // staging col (K)

  for (int k0 = 0; k0 < MDIM; k0 += 32) {
    __syncthreads();                           // protect LDS reuse
    {
      f32x4 f = *reinterpret_cast<const f32x4*>(memB + (size_t)sr * MDIM + k0 + sc);
      u32x2 p;
      p[0] = (unsigned int)f2bf(f[0]) | ((unsigned int)f2bf(f[1]) << 16);
      p[1] = (unsigned int)f2bf(f[2]) | ((unsigned int)f2bf(f[3]) << 16);
      *reinterpret_cast<u32x2*>(&ldsA[sr * 32 + sc]) = p;
    }
    __syncthreads();

    // A fragments (16-bit A layout: lane (h, m=ln); elems 0-7 -> K=8h..8h+7,
    // elems 8-15 -> K=8h+16..8h+23): two ds b128 loads per M-tile.
    union { bf16x16 v; u32x4 q[2]; } ua0, ua1;
    ua0.q[0] = *reinterpret_cast<const u32x4*>(&ldsA[ln * 32 + 8 * h]);
    ua0.q[1] = *reinterpret_cast<const u32x4*>(&ldsA[ln * 32 + 8 * h + 16]);
    ua1.q[0] = *reinterpret_cast<const u32x4*>(&ldsA[(16 + ln) * 32 + 8 * h]);
    ua1.q[1] = *reinterpret_cast<const u32x4*>(&ldsA[(16 + ln) * 32 + 8 * h + 16]);

    #pragma unroll
    for (int j = 0; j < 8; ++j) {
      // B fragment: lane (h, n=ln) holds K=8h..8h+7, 8h+16..8h+23 of column n0+n.
      // Wm is [N,K] row-major -> two runs of 8 contiguous bf16 per lane from global.
      const int n = wave * 128 + j * 16 + ln;
      const unsigned short* bp = wmbf + (size_t)n * MDIM + k0 + 8 * h;
      union { bf16x16 v; u32x4 q[2]; } ub;
      ub.q[0] = *reinterpret_cast<const u32x4*>(bp);
      ub.q[1] = *reinterpret_cast<const u32x4*>(bp + 16);
      acc0[j] = __builtin_amdgcn_wmma_f32_16x16x32_bf16(
          false, ua0.v, false, ub.v, (short)0, acc0[j], false, false);
      acc1[j] = __builtin_amdgcn_wmma_f32_16x16x32_bf16(
          false, ua1.v, false, ub.v, (short)0, acc1[j], false, false);
    }
  }

  // Epilogue: C layout -> row M = mt*16 + g + 8h, col N = wave*128+j*16+ln.
  float ps0[8], ps1[8];
  #pragma unroll
  for (int g = 0; g < 8; ++g) { ps0[g] = 0.0f; ps1[g] = 0.0f; }

  #pragma unroll
  for (int j = 0; j < 8; ++j) {
    const int c = wave * 128 + j * 16 + ln;
    const float qv = qmat[(size_t)b * ADIM + c];
    const float vv = vvec[c];
    #pragma unroll
    for (int g = 0; g < 8; ++g) {
      ps0[g] += vv * fast_tanhf(qv + acc0[j][g]);
      ps1[g] += vv * fast_tanhf(qv + acc1[j][g]);
    }
  }

  // Reduce over the 16 column-lanes within each half-wave.
  #pragma unroll
  for (int g = 0; g < 8; ++g) {
    #pragma unroll
    for (int off = 1; off < 16; off <<= 1) {
      ps0[g] += __shfl_xor(ps0[g], off, 32);
      ps1[g] += __shfl_xor(ps1[g], off, 32);
    }
  }

  if (tid < 32) sred[tid] = 0.0f;
  __syncthreads();
  if (ln == 0) {
    #pragma unroll
    for (int g = 0; g < 8; ++g) {
      atomicAdd(&sred[h * 8 + g], ps0[g]);          // ds_add_f32
      atomicAdd(&sred[16 + h * 8 + g], ps1[g]);
    }
  }
  __syncthreads();
  if (tid < 32)
    scores[(size_t)b * TT + t0 + tid] = sred[tid];
}

// ---------------------------------------------------------------------------
// Kernel 3: masked softmax over T per batch row. grid (BB), 256 threads.
// ---------------------------------------------------------------------------
__global__ __launch_bounds__(256) void softmax_kernel(
    const float* __restrict__ scores, const unsigned char* __restrict__ mask,
    float* __restrict__ attn) {
  __shared__ float red[8];
  const int b    = blockIdx.x;
  const int tid  = threadIdx.x;
  const int wave = tid >> 5;
  const int lane = tid & 31;

  float vals[8];
  float mx = -3.4e38f;
  #pragma unroll
  for (int i = 0; i < 8; ++i) {
    const int t = tid + i * 256;
    float s = scores[(size_t)b * TT + t];
    s = mask[(size_t)b * TT + t] ? s : NEG_INF_F;
    vals[i] = s;
    mx = fmaxf(mx, s);
  }
  for (int off = 16; off > 0; off >>= 1) mx = fmaxf(mx, __shfl_xor(mx, off, 32));
  if (lane == 0) red[wave] = mx;
  __syncthreads();
  #pragma unroll
  for (int w = 0; w < 8; ++w) mx = fmaxf(mx, red[w]);
  __syncthreads();

  float sum = 0.0f;
  #pragma unroll
  for (int i = 0; i < 8; ++i) {
    const float e = __expf(vals[i] - mx);
    vals[i] = e;
    sum += e;
  }
  for (int off = 16; off > 0; off >>= 1) sum += __shfl_xor(sum, off, 32);
  if (lane == 0) red[wave] = sum;
  __syncthreads();
  sum = 0.0f;
  #pragma unroll
  for (int w = 0; w < 8; ++w) sum += red[w];

  const float inv = 1.0f / sum;
  #pragma unroll
  for (int i = 0; i < 8; ++i)
    attn[(size_t)b * TT + tid + i * 256] = vals[i] * inv;
}

// ---------------------------------------------------------------------------
// Kernel 4a: partial context: grid (BB, 16) chunks of 128 tokens, 512 thr.
// ---------------------------------------------------------------------------
__global__ __launch_bounds__(512) void ctx_partial_kernel(
    const float* __restrict__ attn, const float* __restrict__ memory,
    float* __restrict__ part) {
  __shared__ float sa[128];
  const int b = blockIdx.x;
  const int c = blockIdx.y;          // token chunk
  const int d = threadIdx.x;         // 0..511
  if (threadIdx.x < 128)
    sa[threadIdx.x] = attn[(size_t)b * TT + c * 128 + threadIdx.x];
  __syncthreads();
  const float* mp = memory + ((size_t)b * TT + (size_t)c * 128) * MDIM + d;
  float acc = 0.0f;
  #pragma unroll 4
  for (int t = 0; t < 128; ++t) acc += sa[t] * mp[(size_t)t * MDIM];
  part[((size_t)b * 16 + c) * MDIM + d] = acc;
}

// Kernel 4b: deterministic final reduction over the 16 chunks.
__global__ __launch_bounds__(512) void ctx_reduce_kernel(
    const float* __restrict__ part, float* __restrict__ ctx) {
  const int b = blockIdx.x;
  const int d = threadIdx.x;
  float acc = 0.0f;
  #pragma unroll
  for (int c = 0; c < 16; ++c) acc += part[((size_t)b * 16 + c) * MDIM + d];
  ctx[(size_t)b * MDIM + d] = acc;
}

// ---------------------------------------------------------------------------
extern "C" void kernel_launch(void* const* d_in, const int* in_sizes, int n_in,
                              void* d_out, int out_size, void* d_ws, size_t ws_size,
                              hipStream_t stream) {
  const float*         query  = (const float*)d_in[0];          // [32,1024]
  const float*         memory = (const float*)d_in[1];          // [32,2048,512]
  const unsigned char* mask   = (const unsigned char*)d_in[2];  // [32,2048] bool
  const float*         wq     = (const float*)d_in[3];          // [1024,1024]
  const float*         wm     = (const float*)d_in[4];          // [1024,512]
  const float*         vvec   = (const float*)d_in[5];          // [1024]

  float* ctx_out  = (float*)d_out;                   // [32,512]
  float* attn_out = (float*)d_out + BB * MDIM;       // [32,2048]

  // workspace layout (floats): q | scores | ctx-partials | Wm-bf16
  float* ws        = (float*)d_ws;
  float* q_ws      = ws;                             // 32768 f
  float* scores_ws = q_ws + BB * ADIM;               // 65536 f
  float* part_ws   = scores_ws + BB * TT;            // 262144 f
  unsigned short* wmbf = (unsigned short*)(part_ws + BB * 16 * MDIM); // 1MB

  cvt_wm_kernel<<<dim3((ADIM * MDIM) / 256), 256, 0, stream>>>(wm, wmbf, ADIM * MDIM);
  q_kernel<<<dim3(ADIM / 256, BB), 256, 0, stream>>>(query, wq, q_ws);
  score_kernel<<<dim3(TT / 32, BB), 256, 0, stream>>>(memory, wmbf, q_ws, vvec, scores_ws);
  softmax_kernel<<<dim3(BB), 256, 0, stream>>>(scores_ws, mask, attn_out);
  ctx_partial_kernel<<<dim3(BB, 16), 512, 0, stream>>>(attn_out, memory, part_ws);
  ctx_reduce_kernel<<<dim3(BB), 512, 0, stream>>>(part_ws, ctx_out);
}
